// SwinTransformerBlock_33251636805782
// MI455X (gfx1250) — matrix operand order repaired and checked by reading, not compiled, not run on hardware
//
#include <hip/hip_runtime.h>
#include <math.h>

#define HW 56
#define CH 128
#define WSZ 7
#define SHIFT 3
#define NHEAD 4
#define HDIM 32
#define NTOK 49
#define NWIN 64
#define BATCH 64
#define TOKENS (HW*HW)   // 3136

typedef __attribute__((ext_vector_type(16))) _Float16 v16h;
typedef __attribute__((ext_vector_type(8)))  float    v8f;
typedef __attribute__((ext_vector_type(8)))  unsigned v8u;

union Frag16 { v16h h; v8u u; };

__device__ __forceinline__ v8f vzero() {
  v8f z;
#pragma unroll
  for (int e = 0; e < 8; ++e) z[e] = 0.0f;
  return z;
}
__device__ __forceinline__ v8f vsplat(float s) {
  v8f z;
#pragma unroll
  for (int e = 0; e < 8; ++e) z[e] = s;
  return z;
}

// A fragment: 16x32 f16 tile from row-major storage, leading dim ld (halfs).
// Layout used here (contiguous k, same as ldBT/ldBN): lanes 0-15 -> M=lane, lanes 16-31 -> M=lane-16;
// VGPR v holds K pair k,k+1 with k = (lane>>4)*16 + 2v.
__device__ __forceinline__ v16h ldA(const _Float16* p, int ld, int lane) {
  int m = lane & 15, hb = lane >> 4;
  Frag16 f;
#pragma unroll
  for (int v = 0; v < 8; ++v) {
    int k = (hb << 4) + (v << 1);
    f.u[v] = *(const unsigned*)(p + m * ld + k);
  }
  return f.h;
}

// B fragment for X @ W^T: B[k][n] = W[n*ld + k] (weight row n contiguous).
// Layout: lane n = lane&15; lanes 0-15 carry K=0..15, lanes 16-31 K=16..31;
// VGPR v holds pair k = (lane>>4)*16 + 2v.
__device__ __forceinline__ v16h ldBT(const _Float16* p, int ld, int lane) {
  int n = lane & 15, hb = lane >> 4;
  Frag16 f;
#pragma unroll
  for (int v = 0; v < 8; ++v) {
    int k = (hb << 4) + (v << 1);
    f.u[v] = *(const unsigned*)(p + n * ld + k);
  }
  return f.h;
}

// B fragment natural row-major K x N: B[k][n] = p[k*ld + n].
__device__ __forceinline__ v16h ldBN(const _Float16* p, int ld, int lane) {
  int n = lane & 15, hb = lane >> 4;
  v16h b;
#pragma unroll
  for (int v = 0; v < 8; ++v) {
    int k = (hb << 4) + (v << 1);
    b[2 * v]     = p[k * ld + n];
    b[2 * v + 1] = p[(k + 1) * ld + n];
  }
  return b;
}

__device__ __forceinline__ v8f wmma16(v16h a, v16h b, v8f c) {
  return __builtin_amdgcn_wmma_f32_16x16x32_f16(false, a, false, b, (short)0, c,
                                                false, false);
}

// D store: element (m,n): m = v + (lane>>4)*8, n = lane&15.
__device__ __forceinline__ void stD_h(_Float16* p, int ld, int lane, v8f acc) {
  int n = lane & 15, hb = lane >> 4;
#pragma unroll
  for (int v = 0; v < 8; ++v) p[(v + hb * 8) * ld + n] = (_Float16)acc[v];
}
__device__ __forceinline__ void stD_f(float* p, int ld, int lane, v8f acc) {
  int n = lane & 15, hb = lane >> 4;
#pragma unroll
  for (int v = 0; v < 8; ++v) p[(v + hb * 8) * ld + n] = acc[v];
}

// region id along one axis for the shifted-window attention mask
__device__ __forceinline__ int regionOf(int wcoord, int t) {
  int g = wcoord * WSZ + t;
  return (g < HW - WSZ) ? 0 : (g < HW - SHIFT ? 1 : 2);
}

// ---------------------------------------------------------------- weights f32->f16
__global__ void wconvert(const float* __restrict__ qkvw, const float* __restrict__ projw,
                         const float* __restrict__ fc1w, const float* __restrict__ fc2w,
                         _Float16* __restrict__ out) {
  int i = blockIdx.x * 256 + threadIdx.x;
  if (i >= 196608) return;
  float v;
  if (i < 49152)       v = qkvw[i];            // 384x128
  else if (i < 65536)  v = projw[i - 49152];   // 128x128
  else if (i < 131072) v = fc1w[i - 65536];    // 512x128
  else                 v = fc2w[i - 131072];   // 128x512
  out[i] = (_Float16)v;
}

// ---------------------------------------------------------------- fused window attention
// one workgroup (8 wave32) per window; window padded 49 -> 64 tokens
__global__ __launch_bounds__(256) void swin_attn(
    const float* __restrict__ x, const float* __restrict__ n1g,
    const float* __restrict__ n1b, const float* __restrict__ qkvb,
    const float* __restrict__ rpb, const float* __restrict__ projb,
    const _Float16* __restrict__ w16, float* __restrict__ y) {
  extern __shared__ __align__(16) char smem[];
  _Float16* sH   = (_Float16*)smem;              // 64 x 128 f16   (16 KB)  h / attn-out
  _Float16* sQKV = (_Float16*)(smem + 16384);    // 64 x 384 f16   (48 KB)  q|k|v
  float*    sS   = (float*)(smem + 65536);       // 4 x 64 x 64 f32 (64 KB) scores; f16 P overlay
  float*    sRPB = (float*)(smem + 131072);      // 169 x 4 f32     (2.7 KB)

  int wave = threadIdx.x >> 5, lane = threadIdx.x & 31;
  int b = blockIdx.x >> 6;
  int win = blockIdx.x & 63;
  int wr = win >> 3, wc = win & 7;

  for (int i = threadIdx.x; i < 676; i += 256) sRPB[i] = rpb[i];

  // ---- phase 1: LN1 + cyclic shift gather into LDS (f16), zero-pad rows 49..63
  const float* xb = x + (long)b * TOKENS * CH;
  for (int t = wave; t < 64; t += 8) {
    if (t < NTOK) {
      int i = t / WSZ, j = t % WSZ;
      int r  = (wr * WSZ + i + SHIFT) % HW;
      int c2 = (wc * WSZ + j + SHIFT) % HW;
      const float* row = xb + (long)(r * HW + c2) * CH;
      float4 val = ((const float4*)row)[lane];
      float s  = val.x + val.y + val.z + val.w;
      float s2 = val.x * val.x + val.y * val.y + val.z * val.z + val.w * val.w;
#pragma unroll
      for (int off = 16; off > 0; off >>= 1) {
        s  += __shfl_xor(s,  off, 32);
        s2 += __shfl_xor(s2, off, 32);
      }
      float mu = s * (1.0f / CH);
      float rs = rsqrtf(s2 * (1.0f / CH) - mu * mu + 1e-5f);
      const float* vp = (const float*)&val;
#pragma unroll
      for (int q2 = 0; q2 < 4; ++q2) {
        int c = lane * 4 + q2;
        sH[t * CH + c] = (_Float16)((vp[q2] - mu) * rs * n1g[c] + n1b[c]);
      }
    } else {
      unsigned* z = (unsigned*)(sH + t * CH);
      for (int e2 = lane; e2 < 64; e2 += 32) z[e2] = 0u;
    }
  }
  __syncthreads();

  // ---- phase 2: qkv = h @ Wqkv^T + b  (q pre-scaled by 1/sqrt(HD))
  {
    const float qscale = 0.17677669529663689f;
#pragma unroll
    for (int ntl = 0; ntl < 3; ++ntl) {
      int nt = wave * 3 + ntl;                      // 24 col-tiles of 16
      __builtin_prefetch(w16 + nt * 16 * CH, 0, 0);
      float bias = qkvb[nt * 16 + (lane & 15)];
      v8f acc[4];
#pragma unroll
      for (int mt = 0; mt < 4; ++mt) acc[mt] = vsplat(bias);
#pragma unroll
      for (int kk = 0; kk < 4; ++kk) {
        v16h bf = ldBT(w16 + nt * 16 * CH + kk * 32, CH, lane);
#pragma unroll
        for (int mt = 0; mt < 4; ++mt) {
          v16h af = ldA(sH + mt * 16 * CH + kk * 32, CH, lane);
          acc[mt] = wmma16(af, bf, acc[mt]);
        }
      }
      bool isQ = nt < 8;
#pragma unroll
      for (int mt = 0; mt < 4; ++mt) {
        v8f a = acc[mt];
        if (isQ) {
#pragma unroll
          for (int e = 0; e < 8; ++e) a[e] *= qscale;
        }
        stD_h(sQKV + mt * 16 * 384 + nt * 16, 384, lane, a);
      }
    }
  }
  __syncthreads();

  // ---- phase 3: S = q @ k^T per head (K = HD = 32 -> single WMMA per tile)
  {
    int head = wave >> 1;
    int mtBase = (wave & 1) * 2;
#pragma unroll
    for (int nt2 = 0; nt2 < 4; ++nt2) {
      v16h bf = ldBT(sQKV + (nt2 * 16) * 384 + CH + head * 32, 384, lane);
#pragma unroll
      for (int mi = 0; mi < 2; ++mi) {
        int mt = mtBase + mi;
        v16h af = ldA(sQKV + (mt * 16) * 384 + head * 32, 384, lane);
        v8f s = wmma16(af, bf, vzero());
        stD_f(sS + head * 4096 + (mt * 16) * 64 + nt2 * 16, 64, lane, s);
      }
    }
  }
  __syncthreads();

  // ---- phase 4: + rel-pos bias + shift mask, softmax; write f16 P in-place over sS
  {
    int head = threadIdx.x >> 6;
    int n0 = threadIdx.x & 63;         // 4 heads x 64 rows == 256 threads
    float* srow = sS + head * 4096 + n0 * 64;
    _Float16* prow = (_Float16*)srow;  // safe in-place overlay (ascending m)
    if (n0 < NTOK) {
      int i1 = n0 / WSZ, j1 = n0 % WSZ;
      int reg1 = regionOf(wr, i1) * 3 + regionOf(wc, j1);
      float mx = -3.0e38f;
      for (int m = 0; m < NTOK; ++m) {
        int i2 = m / WSZ, j2 = m % WSZ;
        float bias = sRPB[((i1 - i2 + WSZ - 1) * (2 * WSZ - 1) + (j1 - j2 + WSZ - 1)) * NHEAD + head];
        int reg2 = regionOf(wr, i2) * 3 + regionOf(wc, j2);
        float s = srow[m] + bias + ((reg1 == reg2) ? 0.0f : -100.0f);
        srow[m] = s;
        mx = fmaxf(mx, s);
      }
      float sum = 0.0f;
      for (int m = 0; m < NTOK; ++m) {
        float e = __expf(srow[m] - mx);
        srow[m] = e;
        sum += e;
      }
      float inv = 1.0f / sum;
      for (int m = 0; m < NTOK; ++m) prow[m] = (_Float16)(srow[m] * inv);
      for (int m = NTOK; m < 64; ++m) prow[m] = (_Float16)0.0f;
    } else {
      for (int m = 0; m < 64; ++m) prow[m] = (_Float16)0.0f;
    }
  }
  __syncthreads();

  // ---- phase 5: out = P @ v per head (K = 64 -> 2 WMMA) -> sH (64x128 f16)
  {
    int head = wave >> 1, ntv = wave & 1;
    v8f acc[4];
#pragma unroll
    for (int mt = 0; mt < 4; ++mt) acc[mt] = vzero();
#pragma unroll
    for (int kk = 0; kk < 2; ++kk) {
      v16h bf = ldBN(sQKV + (kk * 32) * 384 + 256 + head * 32 + ntv * 16, 384, lane);
#pragma unroll
      for (int mt = 0; mt < 4; ++mt) {
        const _Float16* pp = (const _Float16*)(sS + head * 4096 + (mt * 16) * 64) + kk * 32;
        v16h af = ldA(pp, 128, lane);  // f16 rows overlaid on 64-float rows
        acc[mt] = wmma16(af, bf, acc[mt]);
      }
    }
#pragma unroll
    for (int mt = 0; mt < 4; ++mt)
      stD_h(sH + (mt * 16) * CH + head * 32 + ntv * 16, CH, lane, acc[mt]);
  }
  __syncthreads();

  // ---- phase 6: proj + residual, reverse-shift scatter to y
  {
    const _Float16* Wp = w16 + 49152;
    int nt = wave;  // 8 col-tiles of 16
    __builtin_prefetch(Wp + nt * 16 * CH, 0, 0);
    float pb = projb[nt * 16 + (lane & 15)];
    v8f acc[4];
#pragma unroll
    for (int mt = 0; mt < 4; ++mt) acc[mt] = vsplat(pb);
#pragma unroll
    for (int kk = 0; kk < 4; ++kk) {
      v16h bf = ldBT(Wp + nt * 16 * CH + kk * 32, CH, lane);
#pragma unroll
      for (int mt = 0; mt < 4; ++mt) {
        v16h af = ldA(sH + (mt * 16) * CH + kk * 32, CH, lane);
        acc[mt] = wmma16(af, bf, acc[mt]);
      }
    }
    int n = lane & 15, hb = lane >> 4;
#pragma unroll
    for (int mt = 0; mt < 4; ++mt) {
#pragma unroll
      for (int v = 0; v < 8; ++v) {
        int tok = mt * 16 + v + hb * 8;
        if (tok < NTOK) {
          int i = tok / WSZ, j = tok % WSZ;
          int r  = (wr * WSZ + i + SHIFT) % HW;
          int c2 = (wc * WSZ + j + SHIFT) % HW;
          long off = ((long)b * TOKENS + r * HW + c2) * CH + nt * 16 + n;
          y[off] = x[off] + acc[mt][v];
        }
      }
    }
  }
}

// ---------------------------------------------------------------- fused MLP
// one workgroup per 64 tokens: LN2 -> fc1 + exact GELU -> fc2 -> residual
__global__ __launch_bounds__(256) void swin_mlp(
    const float* __restrict__ y, const float* __restrict__ n2g,
    const float* __restrict__ n2b, const float* __restrict__ fc1b,
    const float* __restrict__ fc2b, const _Float16* __restrict__ w16,
    float* __restrict__ out) {
  extern __shared__ __align__(16) char smem[];
  _Float16* sM = (_Float16*)smem;            // 64 x 128 f16 (16 KB)
  _Float16* sA = (_Float16*)(smem + 16384);  // 64 x 512 f16 (64 KB)
  int wave = threadIdx.x >> 5, lane = threadIdx.x & 31;
  long row0 = (long)blockIdx.x * 64;

  // LN2
  for (int t = wave; t < 64; t += 8) {
    const float* row = y + (row0 + t) * CH;
    float4 val = ((const float4*)row)[lane];
    float s  = val.x + val.y + val.z + val.w;
    float s2 = val.x * val.x + val.y * val.y + val.z * val.z + val.w * val.w;
#pragma unroll
    for (int off = 16; off > 0; off >>= 1) {
      s  += __shfl_xor(s,  off, 32);
      s2 += __shfl_xor(s2, off, 32);
    }
    float mu = s * (1.0f / CH);
    float rs = rsqrtf(s2 * (1.0f / CH) - mu * mu + 1e-5f);
    const float* vp = (const float*)&val;
#pragma unroll
    for (int q2 = 0; q2 < 4; ++q2) {
      int c = lane * 4 + q2;
      sM[t * CH + c] = (_Float16)((vp[q2] - mu) * rs * n2g[c] + n2b[c]);
    }
  }
  __syncthreads();

  // fc1 (128 -> 512) + exact GELU
  {
    const _Float16* W1 = w16 + 65536;
#pragma unroll
    for (int ntl = 0; ntl < 4; ++ntl) {
      int nt = wave * 4 + ntl;  // 32 col-tiles
      __builtin_prefetch(W1 + nt * 16 * CH, 0, 0);
      float b1 = fc1b[nt * 16 + (lane & 15)];
      v8f acc[4];
#pragma unroll
      for (int mt = 0; mt < 4; ++mt) acc[mt] = vsplat(b1);
#pragma unroll
      for (int kk = 0; kk < 4; ++kk) {
        v16h bf = ldBT(W1 + nt * 16 * CH + kk * 32, CH, lane);
#pragma unroll
        for (int mt = 0; mt < 4; ++mt) {
          v16h af = ldA(sM + mt * 16 * CH + kk * 32, CH, lane);
          acc[mt] = wmma16(af, bf, acc[mt]);
        }
      }
#pragma unroll
      for (int mt = 0; mt < 4; ++mt) {
        v8f g;
#pragma unroll
        for (int e = 0; e < 8; ++e) {
          float v = acc[mt][e];
          g[e] = 0.5f * v * (1.0f + erff(v * 0.70710678118654752f));
        }
        stD_h(sA + mt * 16 * 512 + nt * 16, 512, lane, g);
      }
    }
  }
  __syncthreads();

  // fc2 (512 -> 128) + residual
  {
    const _Float16* W2 = w16 + 131072;
    int nt = wave;  // 8 col-tiles
    __builtin_prefetch(W2 + nt * 16 * 512, 0, 0);
    v8f acc[4];
#pragma unroll
    for (int mt = 0; mt < 4; ++mt) acc[mt] = vzero();
#pragma unroll
    for (int kk = 0; kk < 16; ++kk) {
      v16h bf = ldBT(W2 + nt * 16 * 512 + kk * 32, 512, lane);
#pragma unroll
      for (int mt = 0; mt < 4; ++mt) {
        v16h af = ldA(sA + mt * 16 * 512 + kk * 32, 512, lane);
        acc[mt] = wmma16(af, bf, acc[mt]);
      }
    }
    float b2 = fc2b[nt * 16 + (lane & 15)];
    int n = lane & 15, hb = lane >> 4;
#pragma unroll
    for (int mt = 0; mt < 4; ++mt) {
#pragma unroll
      for (int v = 0; v < 8; ++v) {
        long r = row0 + mt * 16 + v + hb * 8;
        long off = r * CH + nt * 16 + n;
        out[off] = y[off] + b2 + acc[mt][v];
      }
    }
  }
}

extern "C" void kernel_launch(void* const* d_in, const int* in_sizes, int n_in,
                              void* d_out, int out_size, void* d_ws, size_t ws_size,
                              hipStream_t stream) {
  (void)in_sizes; (void)n_in; (void)out_size; (void)ws_size;
  const float* x      = (const float*)d_in[0];
  const float* n1g    = (const float*)d_in[1];
  const float* n1b    = (const float*)d_in[2];
  const float* qkv_w  = (const float*)d_in[3];
  const float* qkv_b  = (const float*)d_in[4];
  const float* rpb    = (const float*)d_in[5];
  const float* proj_w = (const float*)d_in[6];
  const float* proj_b = (const float*)d_in[7];
  const float* n2g    = (const float*)d_in[8];
  const float* n2b    = (const float*)d_in[9];
  const float* fc1_w  = (const float*)d_in[10];
  const float* fc1_b  = (const float*)d_in[11];
  const float* fc2_w  = (const float*)d_in[12];
  const float* fc2_b  = (const float*)d_in[13];

  _Float16* w16 = (_Float16*)d_ws;                  // 196608 f16 = 384 KB
  float* y = (float*)((char*)d_ws + 393216);        // residual (B,3136,128) fp32
  float* out = (float*)d_out;

  const int attnLds = 16384 + 49152 + 65536 + 2704; // 133776 B
  const int mlpLds  = 16384 + 65536;                // 81920 B
  (void)hipFuncSetAttribute((const void*)swin_attn,
                            hipFuncAttributeMaxDynamicSharedMemorySize, attnLds);
  (void)hipFuncSetAttribute((const void*)swin_mlp,
                            hipFuncAttributeMaxDynamicSharedMemorySize, mlpLds);

  wconvert<<<768, 256, 0, stream>>>(qkv_w, proj_w, fc1_w, fc2_w, w16);
  swin_attn<<<BATCH * NWIN, 256, attnLds, stream>>>(x, n1g, n1b, qkv_b, rpb,
                                                    proj_b, w16, y);
  swin_mlp<<<BATCH * TOKENS / 64, 256, mlpLds, stream>>>(y, n2g, n2b, fc1_b,
                                                         fc2_b, w16, out);
}
